// Bottle_91242285236866
// MI455X (gfx1250) — matrix operand order repaired and checked by reading, not compiled, not run on hardware
//
#include <hip/hip_runtime.h>
#include <hip/hip_bf16.h>
#include <math.h>

// Problem constants (from reference)
#define N_TOT 131072
#define K_CODES 1024
#define C_DIM 64

// bf16 codebook row stride in LDS (bytes): 64 elems + 8 pad = 72 bf16 = 144 B
// multiple of 16 (b128-aligned); 36-dword lane stride is bank-clean per half.
#define CB_ROW_B 144

typedef __attribute__((ext_vector_type(16))) __bf16 v16bf;
typedef __attribute__((ext_vector_type(8)))  __bf16 v8bf;
typedef __attribute__((ext_vector_type(2)))  __bf16 v2bf;
typedef __attribute__((ext_vector_type(8)))  float  v8f;

// ---------------------------------------------------------------------------
// Kernel 0: zero workspace accumulators (sum_cur[K*C], elem_cur[K], loss[1])
// ---------------------------------------------------------------------------
__global__ __launch_bounds__(256) void vq_init_ws(float* ws, int n) {
    int i = blockIdx.x * 256 + threadIdx.x;
    if (i < n) ws[i] = 0.0f;
}

// ---------------------------------------------------------------------------
// Kernel 1: WMMA distances + argmin + gather/scatter
//   grid = N/128 blocks, 256 threads (8 waves, each wave owns 16 rows)
// ---------------------------------------------------------------------------
__global__ __launch_bounds__(256) void vq_wmma_kernel(
    const float* __restrict__ ze,
    const float* __restrict__ codebook,
    float* __restrict__ out_zq,     // [N*C]
    float* __restrict__ out_zi,     // [N] (as float, per output dtype)
    float* __restrict__ sum_cur,    // [K*C] atomic accum
    float* __restrict__ elem_cur,   // [K]   atomic accum
    float* __restrict__ loss_acc)   // [1]   atomic accum
{
    extern __shared__ char smem[];
    char*  cbf     = smem;                                   // K * 144 B (bf16 rows)
    float* cb2_lds = (float*)(smem + K_CODES * CB_ROW_B);    // K floats
    float* ze_lds  = cb2_lds + K_CODES;                      // 128*C floats
    int*   zi_lds  = (int*)(ze_lds + 128 * C_DIM);           // 128 ints

    const int tid  = threadIdx.x;
    const int wave = tid >> 5;
    const int lane = tid & 31;
    const int half = lane >> 4;   // 0: lanes 0-15, 1: lanes 16-31
    const int nsub = lane & 15;   // A-row / B-col / D-col within half

    // ---- stage codebook as bf16 into LDS (packed cvt, one pass) ----
    #pragma unroll 4
    for (int p = tid; p < K_CODES * C_DIM / 2; p += 256) {
        const int code = p >> 5;
        const int kp   = p & 31;
        const float2 f = ((const float2*)codebook)[p];
        v2bf t;
        t[0] = (__bf16)f.x;
        t[1] = (__bf16)f.y;
        *(v2bf*)(cbf + code * CB_ROW_B + kp * 4) = t;
    }
    // ---- stage this block's 128-row ze tile (fp32) ----
    const int row0 = blockIdx.x * 128;
    {
        const float4* g = (const float4*)(ze + (size_t)row0 * C_DIM);
        float4* s = (float4*)ze_lds;
        #pragma unroll
        for (int i = tid; i < 128 * C_DIM / 4; i += 256) s[i] = g[i];
    }
    __syncthreads();

    // ---- codebook squared norms (from the bf16 values the WMMA multiplies) ----
    for (int k = tid; k < K_CODES; k += 256) {
        const char* row = cbf + k * CB_ROW_B;
        float s = 0.0f;
        #pragma unroll
        for (int j = 0; j < 8; ++j) {
            v8bf v = *(const v8bf*)(row + j * 16);
            #pragma unroll
            for (int e = 0; e < 8; ++e) {
                const float x = (float)v[e];
                s += x * x;
            }
        }
        cb2_lds[k] = s;
    }
    __syncthreads();

    const float* zew = ze_lds + wave * 16 * C_DIM;  // this wave's 16x64 tile

    // ---- build A fragments (contiguous k order, same as B), K-chunks 0 and 32 ----
    // lane half h: K = kc + half*16 + h, row M = nsub
    v16bf a0, a1;
    #pragma unroll
    for (int h = 0; h < 16; ++h) {
        const int kk = half * 16 + h;
        a0[h] = (__bf16)zew[nsub * C_DIM + kk];
        a1[h] = (__bf16)zew[nsub * C_DIM + 32 + kk];
    }

    // ---- argmin over 64 codebook column-blocks of 16 codes ----
    // Key identity: ||ze||^2 is constant per row, so argmin_k dist(row,k)
    //             == argmin_k ( ||cb_k||^2 - 2 <ze_row, cb_k> ).  No ze norms.
    float bestd[8];
    int   besti[8];
    #pragma unroll
    for (int v = 0; v < 8; ++v) { bestd[v] = 3.4e38f; besti[v] = 0; }

    // prologue: block 0
    const char* bp = cbf + nsub * CB_ROW_B + half * 32;
    const float* c2ptr = cb2_lds + nsub;
    v8f acc;
    {
        const v8bf l0 = *(const v8bf*)(bp);
        const v8bf h0 = *(const v8bf*)(bp + 16);
        const v8bf l1 = *(const v8bf*)(bp + 64);
        const v8bf h1 = *(const v8bf*)(bp + 80);
        const v16bf b0 = __builtin_shufflevector(l0, h0,
            0, 1, 2, 3, 4, 5, 6, 7, 8, 9, 10, 11, 12, 13, 14, 15);
        const v16bf b1 = __builtin_shufflevector(l1, h1,
            0, 1, 2, 3, 4, 5, 6, 7, 8, 9, 10, 11, 12, 13, 14, 15);
        v8f z = {};
        z = __builtin_amdgcn_wmma_f32_16x16x32_bf16(
                false, a0, false, b0, (short)0, z, false, false);
        z = __builtin_amdgcn_wmma_f32_16x16x32_bf16(
                false, a1, false, b1, (short)0, z, false, false);
        acc = z;
    }
    float c2p   = c2ptr[0];
    int   codep = nsub;
    bp    += 16 * CB_ROW_B;
    c2ptr += 16;

    // steady state: issue loads(blk) -> consume(blk-1) -> wmma pair(blk)
    // Rolled loop (no unroll!) keeps VGPRs < 128 and folds ds offsets.
    #pragma clang loop unroll(disable)
    for (int blk = 1; blk < K_CODES / 16; ++blk) {
        const v8bf l0 = *(const v8bf*)(bp);
        const v8bf h0 = *(const v8bf*)(bp + 16);
        const v8bf l1 = *(const v8bf*)(bp + 64);
        const v8bf h1 = *(const v8bf*)(bp + 80);
        const float c2n = c2ptr[0];

        // consume previous block while the loads are in flight
        #pragma unroll
        for (int v = 0; v < 8; ++v) {
            const float d = __builtin_fmaf(-2.0f, acc[v], c2p);
            if (d < bestd[v]) { bestd[v] = d; besti[v] = codep; }
        }

        const v16bf b0 = __builtin_shufflevector(l0, h0,
            0, 1, 2, 3, 4, 5, 6, 7, 8, 9, 10, 11, 12, 13, 14, 15);
        const v16bf b1 = __builtin_shufflevector(l1, h1,
            0, 1, 2, 3, 4, 5, 6, 7, 8, 9, 10, 11, 12, 13, 14, 15);
        v8f z = {};
        z = __builtin_amdgcn_wmma_f32_16x16x32_bf16(
                false, a0, false, b0, (short)0, z, false, false);
        z = __builtin_amdgcn_wmma_f32_16x16x32_bf16(
                false, a1, false, b1, (short)0, z, false, false);
        acc = z;

        c2p = c2n;
        codep += 16;
        bp    += 16 * CB_ROW_B;
        c2ptr += 16;
    }
    // epilogue: last block
    #pragma unroll
    for (int v = 0; v < 8; ++v) {
        const float d = __builtin_fmaf(-2.0f, acc[v], c2p);
        if (d < bestd[v]) { bestd[v] = d; besti[v] = codep; }
    }

    // ---- argmin reduce across the 16 lanes of each half (butterfly) ----
    #pragma unroll
    for (int off = 1; off < 16; off <<= 1) {
        #pragma unroll
        for (int v = 0; v < 8; ++v) {
            const float od = __shfl_xor(bestd[v], off, 32);
            const int   oi = __shfl_xor(besti[v], off, 32);
            if (od < bestd[v] || (od == bestd[v] && oi < besti[v])) {
                bestd[v] = od; besti[v] = oi;
            }
        }
    }

    // ---- publish zi (lane nsub==0 of each half holds 8 rows) ----
    // D layout: VGPR v holds row m = v + 8*half
    if (nsub == 0) {
        #pragma unroll
        for (int v = 0; v < 8; ++v) {
            const int m = v + 8 * half;
            zi_lds[wave * 16 + m] = besti[v];
            out_zi[row0 + wave * 16 + m] = (float)besti[v];
        }
    }
    __syncthreads();

    // ---- gather exact fp32 zq from global (codebook is L2-resident),
    //      scatter segment sums via f32 atomics, accumulate commit loss ----
    float lloss = 0.0f;
    for (int r = 0; r < 16; ++r) {
        const int gk = zi_lds[wave * 16 + r];
        const float* q = codebook + (size_t)gk * C_DIM;
        const float* z = zew + r * C_DIM;
        const float q0 = q[lane], q1 = q[lane + 32];
        const float z0 = z[lane], z1 = z[lane + 32];
        const size_t gr = (size_t)(row0 + wave * 16 + r) * C_DIM;
        out_zq[gr + lane]      = q0;
        out_zq[gr + lane + 32] = q1;
        atomicAdd(&sum_cur[gk * C_DIM + lane],      z0);
        atomicAdd(&sum_cur[gk * C_DIM + lane + 32], z1);
        const float d0 = z0 - q0, d1 = z1 - q1;
        lloss += d0 * d0 + d1 * d1;
        if (lane == 0) atomicAdd(&elem_cur[gk], 1.0f);
    }
    #pragma unroll
    for (int off = 16; off >= 1; off >>= 1) lloss += __shfl_xor(lloss, off, 32);
    if (lane == 0) atomicAdd(loss_acc, lloss);
}

// ---------------------------------------------------------------------------
// Kernel 2: EMA update + entropy/usage reductions + cb_updated + scalars
// ---------------------------------------------------------------------------
__global__ __launch_bounds__(1024) void vq_finalize(
    const float* __restrict__ codebook_sum,
    const float* __restrict__ codebook_elem,
    const float* __restrict__ codebook_random,
    const float* __restrict__ sum_cur,
    const float* __restrict__ elem_cur,
    const float* __restrict__ loss_acc,
    float* __restrict__ out_loss,
    float* __restrict__ out_usage,
    float* __restrict__ out_entropy,
    float* __restrict__ out_cb)
{
    __shared__ float red[1024];
    const int t = threadIdx.x;
    const float mu = 0.99f, one_mu = 0.01f;

    const float ne = mu * codebook_elem[t] + one_mu * elem_cur[t];
    const float usage = (ne >= 1.0f) ? 1.0f : 0.0f;

    // total = new_elem.sum()
    red[t] = ne; __syncthreads();
    #pragma unroll
    for (int s = 512; s > 0; s >>= 1) {
        if (t < s) red[t] += red[t + s];
        __syncthreads();
    }
    const float total = red[0]; __syncthreads();

    // entropy = -sum(p * log(p + 1e-8))
    const float p = ne / total;
    red[t] = -p * logf(p + 1e-8f); __syncthreads();
    #pragma unroll
    for (int s = 512; s > 0; s >>= 1) {
        if (t < s) red[t] += red[t + s];
        __syncthreads();
    }
    const float entropy = red[0]; __syncthreads();

    // usage.sum()
    red[t] = usage; __syncthreads();
    #pragma unroll
    for (int s = 512; s > 0; s >>= 1) {
        if (t < s) red[t] += red[t + s];
        __syncthreads();
    }
    const float usage_sum = red[0];

    // cb_updated row
    const float inv_ne = 1.0f / ne;
    #pragma unroll 8
    for (int c = 0; c < C_DIM; ++c) {
        const float ns = mu * codebook_sum[t * C_DIM + c]
                       + one_mu * sum_cur[t * C_DIM + c];
        const float cn = ns * inv_ne;
        out_cb[t * C_DIM + c] =
            usage * cn + (1.0f - usage) * codebook_random[t * C_DIM + c];
    }

    if (t == 0) {
        out_loss[0]    = loss_acc[0] / (float)((size_t)N_TOT * C_DIM);
        out_usage[0]   = usage_sum;
        out_entropy[0] = entropy;
    }
}

// ---------------------------------------------------------------------------
// Host-side launch
// ---------------------------------------------------------------------------
extern "C" void kernel_launch(void* const* d_in, const int* in_sizes, int n_in,
                              void* d_out, int out_size, void* d_ws, size_t ws_size,
                              hipStream_t stream) {
    const float* ze              = (const float*)d_in[0];
    const float* codebook        = (const float*)d_in[1];
    const float* codebook_sum    = (const float*)d_in[2];
    const float* codebook_elem   = (const float*)d_in[3];
    const float* codebook_random = (const float*)d_in[4];

    // outputs flat in return order: zq_st[N*C], zi[N], loss, usage_sum,
    // entropy, cb_updated[K*C]
    float* out         = (float*)d_out;
    float* out_zq      = out;
    float* out_zi      = out + (size_t)N_TOT * C_DIM;
    float* out_loss    = out_zi + N_TOT;
    float* out_usage   = out_loss + 1;
    float* out_entropy = out_usage + 1;
    float* out_cb      = out_entropy + 1;

    // workspace: sum_cur[K*C], elem_cur[K], loss_acc[1]
    float* sum_cur  = (float*)d_ws;
    float* elem_cur = sum_cur + (size_t)K_CODES * C_DIM;
    float* loss_acc = elem_cur + K_CODES;
    const int ws_elems = K_CODES * C_DIM + K_CODES + 1;

    vq_init_ws<<<(ws_elems + 255) / 256, 256, 0, stream>>>((float*)d_ws, ws_elems);

    // dynamic LDS: bf16 codebook (fragment layout) + norms + 128-row ze tile + zi
    const size_t shmem = (size_t)K_CODES * CB_ROW_B                 // bf16 codebook
                       + (size_t)K_CODES * sizeof(float)            // cb2
                       + (size_t)128 * C_DIM * sizeof(float)        // ze tile
                       + 128 * sizeof(int);                         // zi tile
    vq_wmma_kernel<<<N_TOT / 128, 256, shmem, stream>>>(
        ze, codebook, out_zq, out_zi, sum_cur, elem_cur, loss_acc);

    vq_finalize<<<1, 1024, 0, stream>>>(
        codebook_sum, codebook_elem, codebook_random,
        sum_cur, elem_cur, loss_acc,
        out_loss, out_usage, out_entropy, out_cb);
}
